// PinwheelAttnBlock_76579266888021
// MI455X (gfx1250) — matrix operand order repaired and checked by reading, not compiled, not run on hardware
//
#include <hip/hip_runtime.h>

// ---------------------------------------------------------------------------
// PinwheelAttnBlock fused pipeline for MI455X (gfx1250, wave32, WMMA).
//
// Bandwidth-bound (~68 GFLOP vs ~105MB fp32 activations): f16 intermediates
// (L2-resident), WMMA f16->f32 GEMMs for all 1x1 convs, fused bias+SiLU,
// attention combine fused into the f1 GEMM's X-tile loader.
// K-loop is software-pipelined: next slice's global_load_b128s are in flight
// during the current slice's WMMA phase. A-tile rows are address-clamped
// (not predicated) since out-of-range rows are never stored.
// ---------------------------------------------------------------------------

typedef __attribute__((ext_vector_type(16))) _Float16 v16h;
typedef __attribute__((ext_vector_type(8)))  _Float16 v8h;
typedef __attribute__((ext_vector_type(4)))  _Float16 v4h;
typedef __attribute__((ext_vector_type(8)))  float    v8f;

#define CCH 256   // channels
#define BB  16    // batch
#define HH  80
#define WWD 80
#define PP  6400  // H*W

__device__ __forceinline__ float silu_f(float g) {
    return g / (1.0f + __expf(-g));
}

__device__ __forceinline__ v16h cat16(v8h lo, v8h hi) {
    v16h r;
#pragma unroll
    for (int i = 0; i < 8; ++i) { r[i] = lo[i]; r[i + 8] = hi[i]; }
    return r;
}

__device__ __forceinline__ v4h cvt4(float4 f) {
    v4h r;
    r[0] = (_Float16)f.x; r[1] = (_Float16)f.y;
    r[2] = (_Float16)f.z; r[3] = (_Float16)f.w;
    return r;
}

enum { XM_F32 = 0, XM_F16 = 1, XM_ATT = 2 };     // X-tile source
enum { OM_SPLITQ = 0, OM_F16 = 1, OM_F32 = 2 };  // output routing

// D[M x 6400] = silu(W[M x 256] @ X[256 x 6400] + bias) per batch image.
// Block: 256 threads = 8 waves -> 64(M) x 64(N) tile; K staged in 32-slices.
template <int XM, int OM>
__global__ void __launch_bounds__(256)
conv1x1_silu_wmma(const float* __restrict__ Wt, const float* __restrict__ Bs,
                  const float* __restrict__ Xf32, const _Float16* __restrict__ Xf16,
                  const _Float16* __restrict__ Val, const float* __restrict__ CtxW,
                  const float* __restrict__ CtxH,
                  float* __restrict__ Qout, _Float16* __restrict__ O16,
                  float* __restrict__ O32, int Mtot)
{
    __shared__ _Float16 As[64][40];  // [m][k]   row stride 80B (16B-aligned)
    __shared__ _Float16 Xs[64][40];  // [n][k]   transposed X tile

    const int b  = blockIdx.z;
    const int N0 = blockIdx.x * 64;
    const int M0 = blockIdx.y * 64;

    const int lane  = threadIdx.x & 31;
    const int wv    = threadIdx.x >> 5;
    const int m16   = wv & 3;    // which 16-row M tile (4 per block)
    const int npair = wv >> 2;   // which 32-col N half (2 per block)
    const int hs    = lane >> 4; // half-wave select
    const int ln    = lane & 15;

    // loader coordinates (fixed per thread)
    const int lm  = threadIdx.x >> 3;          // A: row 0..31 (and +32)
    const int lkk = (threadIdx.x & 7) * 4;     // A: k chunk
    const int lk  = threadIdx.x >> 4;          // X: k 0..15 (and +16)
    const int ln4 = (threadIdx.x & 15) * 4;    // X: pixel chunk

    // A rows: clamp instead of predicate -- out-of-range rows are dead values
    // (their D rows are never stored), so straight-line unconditional loads.
    int Ra = M0 + lm;      if (Ra > Mtot - 1) Ra = Mtot - 1;
    int Rb = M0 + lm + 32; if (Rb > Mtot - 1) Rb = Mtot - 1;

    const size_t bbase = (size_t)b * CCH * PP;
    const int pX = N0 + ln4;                   // X loader pixel
    const int hX = pX / WWD, wX = pX % WWD;    // chunk never crosses a W-row

    v8f acc0 = {};
    v8f acc1 = {};

    // registers holding the next K-slice's global data
    float4 r_wa, r_wb;          // A tile
    float4 r_xa, r_xb;          // XM_F32 / XM_ATT
    v4h    r_h1, r_h2;          // XM_F16
    v4h    r_va, r_vb;          // ATT value
    float4 r_ch1, r_ch2;        // ATT ctx_h
    float  r_cw1, r_cw2;        // ATT ctx_w

    auto prefetch = [&](int K0) {
        r_wa = *(const float4*)&Wt[(size_t)Ra * CCH + K0 + lkk];
        r_wb = *(const float4*)&Wt[(size_t)Rb * CCH + K0 + lkk];
        const int c1 = K0 + lk, c2 = c1 + 16;
        if (XM == XM_F32) {
            r_xa = *(const float4*)&Xf32[bbase + (size_t)c1 * PP + pX];
            r_xb = *(const float4*)&Xf32[bbase + (size_t)c2 * PP + pX];
        } else if (XM == XM_F16) {
            r_h1 = *(const v4h*)&Xf16[bbase + (size_t)c1 * PP + pX];
            r_h2 = *(const v4h*)&Xf16[bbase + (size_t)c2 * PP + pX];
        } else {
            r_xa  = *(const float4*)&Xf32[bbase + (size_t)c1 * PP + pX];
            r_xb  = *(const float4*)&Xf32[bbase + (size_t)c2 * PP + pX];
            r_va  = *(const v4h*)  &Val [bbase + (size_t)c1 * PP + pX];
            r_vb  = *(const v4h*)  &Val [bbase + (size_t)c2 * PP + pX];
            r_ch1 = *(const float4*)&CtxH[((size_t)b * CCH + c1) * WWD + wX];
            r_ch2 = *(const float4*)&CtxH[((size_t)b * CCH + c2) * WWD + wX];
            r_cw1 = CtxW[((size_t)b * CCH + c1) * HH + hX] * 0.0625f;
            r_cw2 = CtxW[((size_t)b * CCH + c2) * HH + hX] * 0.0625f;
        }
    };

    auto commit = [&]() {
        *(v4h*)&As[lm][lkk]      = cvt4(r_wa);
        *(v4h*)&As[lm + 32][lkk] = cvt4(r_wb);
        v4h h1, h2;
        if (XM == XM_F32) {
            h1 = cvt4(r_xa); h2 = cvt4(r_xb);
        } else if (XM == XM_F16) {
            h1 = r_h1; h2 = r_h2;
        } else {
            h1[0] = (_Float16)(fmaf((float)r_va[0], r_cw1 * r_ch1.x, r_xa.x));
            h1[1] = (_Float16)(fmaf((float)r_va[1], r_cw1 * r_ch1.y, r_xa.y));
            h1[2] = (_Float16)(fmaf((float)r_va[2], r_cw1 * r_ch1.z, r_xa.z));
            h1[3] = (_Float16)(fmaf((float)r_va[3], r_cw1 * r_ch1.w, r_xa.w));
            h2[0] = (_Float16)(fmaf((float)r_vb[0], r_cw2 * r_ch2.x, r_xb.x));
            h2[1] = (_Float16)(fmaf((float)r_vb[1], r_cw2 * r_ch2.y, r_xb.y));
            h2[2] = (_Float16)(fmaf((float)r_vb[2], r_cw2 * r_ch2.z, r_xb.z));
            h2[3] = (_Float16)(fmaf((float)r_vb[3], r_cw2 * r_ch2.w, r_xb.w));
        }
#pragma unroll
        for (int i = 0; i < 4; ++i) {
            Xs[ln4 + i][lk]      = h1[i];
            Xs[ln4 + i][lk + 16] = h2[i];
        }
    };

    prefetch(0);

    for (int K0 = 0; K0 < CCH; K0 += 32) {
        __syncthreads();        // previous fragment reads done
        commit();               // waits on in-flight loads land here
        __syncthreads();
        if (K0 + 32 < CCH) prefetch(K0 + 32);  // overlap with WMMA phase

        // ---- fragments (two ds_load_b128 each) ----------------------------
        // A (16x32 f16): lane = row m; K = 16*hs + e (contiguous in As[m][.])
        v8h alo = *(const v8h*)&As[m16 * 16 + ln][hs * 16];
        v8h ahi = *(const v8h*)&As[m16 * 16 + ln][hs * 16 + 8];
        v16h a  = cat16(alo, ahi);

        // B (32x16 f16): lane = col n; K = 16*hs + e (contiguous in Xs[n][.])
        int col0 = npair * 32 + ln;
        v8h b0lo = *(const v8h*)&Xs[col0][hs * 16];
        v8h b0hi = *(const v8h*)&Xs[col0][hs * 16 + 8];
        v16h b0  = cat16(b0lo, b0hi);
        v8h b1lo = *(const v8h*)&Xs[col0 + 16][hs * 16];
        v8h b1hi = *(const v8h*)&Xs[col0 + 16][hs * 16 + 8];
        v16h b1  = cat16(b1lo, b1hi);

        acc0 = __builtin_amdgcn_wmma_f32_16x16x32_f16(false, a, false, b0,
                                                      (short)0, acc0, false, false);
        acc1 = __builtin_amdgcn_wmma_f32_16x16x32_f16(false, a, false, b1,
                                                      (short)0, acc1, false, false);
    }

    // ---- epilogue: bias + SiLU + routed store ------------------------------
    const int p0 = N0 + npair * 32 + ln;
    const int p1 = p0 + 16;
#pragma unroll
    for (int v = 0; v < 8; ++v) {
        int R = M0 + m16 * 16 + hs * 8 + v;  // C/D layout: VGPR v -> M = v+8*hs
        if (R >= Mtot) continue;
        float bv = Bs[R];
        float y0 = silu_f(acc0[v] + bv);
        float y1 = silu_f(acc1[v] + bv);
        if (OM == OM_SPLITQ) {
            if (R == 0) {  // q channel -> fp32 buffer for stable softmax
                Qout[(size_t)b * PP + p0] = y0;
                Qout[(size_t)b * PP + p1] = y1;
            } else {       // k channels -> f16
                size_t base = ((size_t)b * CCH + (R - 1)) * PP;
                O16[base + p0] = (_Float16)y0;
                O16[base + p1] = (_Float16)y1;
            }
        } else if (OM == OM_F16) {
            size_t base = ((size_t)b * CCH + R) * PP;
            O16[base + p0] = (_Float16)y0;
            O16[base + p1] = (_Float16)y1;
        } else {
            size_t base = ((size_t)b * CCH + R) * PP;
            O32[base + p0] = y0;
            O32[base + p1] = y1;
        }
    }
}

// ctx_w: per (b,h): softmax over q_w[b,h,:] (len 80), then
//        ctx_w[b,c,h] = sum_w k_w[b,c,h,w] * score[w]
__global__ void __launch_bounds__(256)
ctx_w_kernel(const float* __restrict__ q, const _Float16* __restrict__ kw,
             float* __restrict__ ctxw)
{
    const int h = blockIdx.x, b = blockIdx.y;
    __shared__ float s[WWD];
    __shared__ float inv_sum;
    const int t = threadIdx.x;
    const float* qr = q + ((size_t)b * HH + h) * WWD;
    if (t < WWD) s[t] = qr[t];
    __syncthreads();
    if (t == 0) {
        float mx = s[0];
        for (int w = 1; w < WWD; ++w) mx = fmaxf(mx, s[w]);
        float sum = 0.0f;
        for (int w = 0; w < WWD; ++w) { float e = __expf(s[w] - mx); s[w] = e; sum += e; }
        inv_sum = 1.0f / sum;
    }
    __syncthreads();
    const int c = t;  // 256 threads = 256 channels
    const _Float16* kr = kw + (((size_t)b * CCH + c) * HH + h) * WWD;
    float acc = 0.0f;
    for (int w = 0; w < WWD; ++w) acc += (float)kr[w] * s[w];
    ctxw[((size_t)b * CCH + c) * HH + h] = acc * inv_sum;
}

// ctx_h: per (b,w): softmax over q_h[b,:,w] (len 80, stride 80), then
//        ctx_h[b,c,w] = sum_h k_h[b,c,h,w] * score[h]
__global__ void __launch_bounds__(256)
ctx_h_kernel(const float* __restrict__ q, const _Float16* __restrict__ kh,
             float* __restrict__ ctxh)
{
    const int w = blockIdx.x, b = blockIdx.y;
    __shared__ float s[HH];
    __shared__ float inv_sum;
    const int t = threadIdx.x;
    const float* qc = q + (size_t)b * PP + w;
    if (t < HH) s[t] = qc[(size_t)t * WWD];
    __syncthreads();
    if (t == 0) {
        float mx = s[0];
        for (int h = 1; h < HH; ++h) mx = fmaxf(mx, s[h]);
        float sum = 0.0f;
        for (int h = 0; h < HH; ++h) { float e = __expf(s[h] - mx); s[h] = e; sum += e; }
        inv_sum = 1.0f / sum;
    }
    __syncthreads();
    const int c = t;
    const _Float16* kr = kh + ((size_t)b * CCH + c) * PP + w;
    float acc = 0.0f;
    for (int h = 0; h < HH; ++h) acc += (float)kr[(size_t)h * WWD] * s[h];
    ctxh[((size_t)b * CCH + c) * WWD + w] = acc * inv_sum;
}

// Depthwise 3x3 (SAME, zero pad) + SiLU, f16 in -> f16 out. Memory-bound.
__global__ void __launch_bounds__(256)
dw3x3_silu_kernel(const _Float16* __restrict__ src, const float* __restrict__ w,
                  const float* __restrict__ bias, _Float16* __restrict__ dst)
{
    size_t i = (size_t)blockIdx.x * 256 + threadIdx.x;  // over B*C*P
    int p = (int)(i % PP);
    size_t bc = i / PP;
    int c = (int)(bc % CCH);
    int h = p / WWD, x = p % WWD;
    float acc = bias[c];
    const float* wc = w + (size_t)c * 9;
    const _Float16* sp = src + bc * PP;
#pragma unroll
    for (int di = -1; di <= 1; ++di) {
#pragma unroll
        for (int dj = -1; dj <= 1; ++dj) {
            int hh = h + di, ww = x + dj;
            if (hh >= 0 && hh < HH && ww >= 0 && ww < WWD)
                acc += (float)sp[hh * WWD + ww] * wc[(di + 1) * 3 + (dj + 1)];
        }
    }
    dst[i] = (_Float16)silu_f(acc);
}

extern "C" void kernel_launch(void* const* d_in, const int* in_sizes, int n_in,
                              void* d_out, int out_size, void* d_ws, size_t ws_size,
                              hipStream_t stream)
{
    (void)in_sizes; (void)n_in; (void)out_size; (void)ws_size;

    const float* x     = (const float*)d_in[0];
    const float* qkw_w = (const float*)d_in[1];
    const float* qkw_b = (const float*)d_in[2];
    const float* qkh_w = (const float*)d_in[3];
    const float* qkh_b = (const float*)d_in[4];
    const float* v_w   = (const float*)d_in[5];
    const float* v_b   = (const float*)d_in[6];
    const float* f1_w  = (const float*)d_in[7];
    const float* f1_b  = (const float*)d_in[8];
    const float* dw_w  = (const float*)d_in[9];
    const float* dw_b  = (const float*)d_in[10];
    const float* f3_w  = (const float*)d_in[11];
    const float* f3_b  = (const float*)d_in[12];
    float* out = (float*)d_out;

    // ---- workspace layout (256B aligned) ----------------------------------
    char* ws = (char*)d_ws;
    size_t off = 0;
    auto alloc = [&](size_t bytes) { size_t o = off; off += (bytes + 255) & ~(size_t)255; return o; };
    const size_t SMALL = (size_t)BB * PP * sizeof(float);          // 409,600 B
    const size_t CTX   = (size_t)BB * CCH * HH * sizeof(float);    // 1,310,720 B
    const size_t BIG   = (size_t)BB * CCH * PP * sizeof(_Float16); // 52,428,800 B

    float*    qw   = (float*)   (ws + alloc(SMALL));
    float*    qh   = (float*)   (ws + alloc(SMALL));
    float*    ctxw = (float*)   (ws + alloc(CTX));
    float*    ctxh = (float*)   (ws + alloc(CTX));
    _Float16* kw   = (_Float16*)(ws + alloc(BIG));
    _Float16* kh   = (_Float16*)(ws + alloc(BIG));
    _Float16* val  = (_Float16*)(ws + alloc(BIG));
    _Float16* y1   = kw;  // dead after ctx_w_kernel
    _Float16* y2   = kh;  // dead after ctx_h_kernel

    const dim3 blk(256);
    const dim3 g256(PP / 64, (CCH + 63) / 64, BB);       // 100 x 4 x 16
    const dim3 g257(PP / 64, (CCH + 1 + 63) / 64, BB);   // 100 x 5 x 16

    // 1) qk_w = silu(qkw @ x + b): row0 -> qw (f32), rows1..256 -> kw (f16)
    conv1x1_silu_wmma<XM_F32, OM_SPLITQ><<<g257, blk, 0, stream>>>(
        qkw_w, qkw_b, x, nullptr, nullptr, nullptr, nullptr, qw, kw, nullptr, CCH + 1);
    // 2) qk_h likewise
    conv1x1_silu_wmma<XM_F32, OM_SPLITQ><<<g257, blk, 0, stream>>>(
        qkh_w, qkh_b, x, nullptr, nullptr, nullptr, nullptr, qh, kh, nullptr, CCH + 1);
    // 3) value = silu(v_w @ x + b) -> f16
    conv1x1_silu_wmma<XM_F32, OM_F16><<<g256, blk, 0, stream>>>(
        v_w, v_b, x, nullptr, nullptr, nullptr, nullptr, nullptr, val, nullptr, CCH);
    // 4/5) axis softmax + weighted reductions
    ctx_w_kernel<<<dim3(HH, BB), blk, 0, stream>>>(qw, kw, ctxw);
    ctx_h_kernel<<<dim3(WWD, BB), blk, 0, stream>>>(qh, kh, ctxh);
    // 6) y1 = silu(f1 @ (val*ctxw*ctxh/16 + x) + b)  (attention fused in loader)
    conv1x1_silu_wmma<XM_ATT, OM_F16><<<g256, blk, 0, stream>>>(
        f1_w, f1_b, x, nullptr, val, ctxw, ctxh, nullptr, y1, nullptr, CCH);
    // 7) y2 = silu(dwconv3x3(y1))
    dw3x3_silu_kernel<<<dim3((unsigned)((size_t)BB * CCH * PP / 256)), blk, 0, stream>>>(
        y1, dw_w, dw_b, y2);
    // 8) out = silu(f3 @ y2 + b) -> fp32
    conv1x1_silu_wmma<XM_F16, OM_F32><<<g256, blk, 0, stream>>>(
        f3_w, f3_b, nullptr, y2, nullptr, nullptr, nullptr, nullptr, nullptr, out, CCH);
}
